// PytorchGeometricGAT_18691697672408
// MI455X (gfx1250) — matrix operand order repaired and checked by reading, not compiled, not run on hardware
//
#include <hip/hip_runtime.h>
#include <hip/hip_bf16.h>

typedef __attribute__((ext_vector_type(16))) _Float16 v16h;
typedef __attribute__((ext_vector_type(8)))  _Float16 v8h;
typedef __attribute__((ext_vector_type(8)))  float    v8f;

#define SLOPE 0.2f

// ---------- monotone float<->uint encoding for atomic max ----------
__device__ __forceinline__ unsigned enc_f(float f) {
    unsigned u = __float_as_uint(f);
    return (u & 0x80000000u) ? ~u : (u | 0x80000000u);
}
__device__ __forceinline__ float dec_f(unsigned u) {
    return (u & 0x80000000u) ? __uint_as_float(u & 0x7FFFFFFFu)
                             : __uint_as_float(~u);
}

// ---------- utility kernels ----------
__global__ void zero_f32(float* p, size_t n) {
    size_t t = (size_t)blockIdx.x * blockDim.x + threadIdx.x;
    if (t < n) p[t] = 0.0f;
}

// W (K x N, f32 row-major)  ->  WT (N x K, f16 row-major)
__global__ void cvt_transpose(const float* __restrict__ W, _Float16* __restrict__ WT,
                              int K, int N) {
    int idx = blockIdx.x * blockDim.x + threadIdx.x;
    if (idx >= K * N) return;
    int k = idx / N, n = idx % N;
    WT[(size_t)n * K + k] = (_Float16)W[(size_t)k * N + n];
}

__global__ void cvt_f16(const float* __restrict__ in, _Float16* __restrict__ out, size_t n) {
    size_t t = (size_t)blockIdx.x * blockDim.x + threadIdx.x;
    if (t < n) out[t] = (_Float16)in[t];
}

// ---------- WMMA GEMM:  C(MxN,f32) = A(MxK,f16 row-major) * BT(NxK,f16 row-major)^T
// Register-blocked: one wave computes a 16 x (16*NT) strip. The A fragment is
// loaded once per k-step and reused by NT B fragments (NT wmma per k-step).
template <int NT>
__global__ void wmma_gemm(const _Float16* __restrict__ A,
                          const _Float16* __restrict__ BT,
                          float* __restrict__ C,
                          int M, int N, int K) {
    int wave   = blockIdx.x * (blockDim.x >> 5) + (threadIdx.x >> 5);
    int tilesN = N / (16 * NT);
    int tileM  = wave / tilesN;
    int tileN0 = (wave % tilesN) * NT;
    if (tileM * 16 >= M) return;          // wave-uniform guard: EXEC stays all-1s

    int lane = threadIdx.x & 31;
    int half = lane >> 4;                 // 0: lanes 0-15, 1: lanes 16-31
    int lrow = lane & 15;

    const _Float16* arow = A + (size_t)(tileM * 16 + lrow) * K;
    const _Float16* brow[NT];
#pragma unroll
    for (int j = 0; j < NT; ++j)
        brow[j] = BT + (size_t)((tileN0 + j) * 16 + lrow) * K;

    v8f acc[NT];
#pragma unroll
    for (int j = 0; j < NT; ++j) acc[j] = (v8f){};

    for (int k0 = 0; k0 < K; k0 += 32) {
        // A fragment: contiguous K order matching B.
        //   elems 0..15 = K = k0 + half*16 + i
        int ka = k0 + half * 16;
        v16h a = *(const v16h*)(arow + ka);

        // B fragments: rows striped across lanes; lane half selects K 0-15/16-31,
        // contiguous in each BT row (32 bytes per fragment per lane).
#pragma unroll
        for (int j = 0; j < NT; ++j) {
            v16h b = *(const v16h*)(brow[j] + k0 + half * 16);
            acc[j] = __builtin_amdgcn_wmma_f32_16x16x32_f16(
                         false, a, false, b, (short)0, acc[j], false, false);
        }
    }

    // D layout: VGPR v -> row = v + half*8, col = lrow
#pragma unroll
    for (int j = 0; j < NT; ++j) {
#pragma unroll
        for (int v = 0; v < 8; ++v) {
            int row = tileM * 16 + v + half * 8;
            int col = (tileN0 + j) * 16 + lrow;
            C[(size_t)row * N + col] = acc[j][v];
        }
    }
}

// ---------- attention logits: a_s[n,h] = <h[n,h,:], att_src[h,:]>, same for dst
__global__ void att_dots(const float* __restrict__ Hf,
                         const float* __restrict__ att_s,
                         const float* __restrict__ att_d,
                         float* __restrict__ as_, float* __restrict__ ad_,
                         int Nn, int H, int Cc) {
    int t = blockIdx.x * blockDim.x + threadIdx.x;
    if (t >= Nn * H) return;
    int n = t / H, h = t % H;
    const float* hp = Hf + (size_t)n * H * Cc + (size_t)h * Cc;
    const float* sp = att_s + (size_t)h * Cc;
    const float* dp = att_d + (size_t)h * Cc;
    float s = 0.0f, d = 0.0f;
    for (int c = 0; c < Cc; ++c) { float v = hp[c]; s += v * sp[c]; d += v * dp[c]; }
    as_[t] = s; ad_[t] = d;
}

__device__ __forceinline__ void edge_sd(const int* ei, int E, int e, int& s, int& d) {
    if (e < E) { s = ei[e]; d = ei[E + e]; } else { s = e - E; d = e - E; }
}

// pass 1: segment max of leaky_relu(a_s[src]+a_d[dst])
__global__ void edge_max(const int* __restrict__ ei, int E, int ET,
                         const float* __restrict__ as_, const float* __restrict__ ad_,
                         unsigned* __restrict__ amax, int H) {
    int e = blockIdx.x * blockDim.x + threadIdx.x;
    if (e >= ET) return;
    int s, d; edge_sd(ei, E, e, s, d);
    for (int h = 0; h < H; ++h) {
        float v = as_[(size_t)s * H + h] + ad_[(size_t)d * H + h];
        v = v > 0.0f ? v : SLOPE * v;
        atomicMax(&amax[(size_t)d * H + h], enc_f(v));
    }
}

// pass 2: ex = exp(e - max), denom[dst] += ex   (ex cached for pass 3)
__global__ void edge_expsum(const int* __restrict__ ei, int E, int ET,
                            const float* __restrict__ as_, const float* __restrict__ ad_,
                            const unsigned* __restrict__ amax,
                            float* __restrict__ ex, float* __restrict__ denom, int H) {
    int e = blockIdx.x * blockDim.x + threadIdx.x;
    if (e >= ET) return;
    int s, d; edge_sd(ei, E, e, s, d);
    for (int h = 0; h < H; ++h) {
        float v = as_[(size_t)s * H + h] + ad_[(size_t)d * H + h];
        v = v > 0.0f ? v : SLOPE * v;
        float m  = dec_f(amax[(size_t)d * H + h]);
        float xv = __expf(v - m);
        ex[(size_t)e * H + h] = xv;
        atomicAdd(&denom[(size_t)d * H + h], xv);
    }
}

// pass 3 (layer 1, 4 heads x 64 ch): wave per edge, 8 channels per lane
__global__ void edge_aggr_l1(const int* __restrict__ ei, int E, int ET,
                             const float* __restrict__ h1,
                             const float* __restrict__ ex,
                             const float* __restrict__ denom,
                             float* __restrict__ out) {
    int e = blockIdx.x * (blockDim.x >> 5) + (threadIdx.x >> 5);
    if (e >= ET) return;
    int lane = threadIdx.x & 31;
    int s, d; edge_sd(ei, E, e, s, d);
    int c0 = lane * 8;
    int h  = c0 >> 6;                     // 64 channels per head
    float alpha = ex[(size_t)e * 4 + h] / (denom[(size_t)d * 4 + h] + 1e-16f);
    const float4* hp = (const float4*)(h1 + (size_t)s * 256 + c0);
    float4 v0 = hp[0], v1 = hp[1];
    float* op = out + (size_t)d * 256 + c0;
    atomicAdd(op + 0, alpha * v0.x); atomicAdd(op + 1, alpha * v0.y);
    atomicAdd(op + 2, alpha * v0.z); atomicAdd(op + 3, alpha * v0.w);
    atomicAdd(op + 4, alpha * v1.x); atomicAdd(op + 5, alpha * v1.y);
    atomicAdd(op + 6, alpha * v1.z); atomicAdd(op + 7, alpha * v1.w);
}

// pass 3 (layer 2, 1 head x 16 ch): thread per (edge, channel)
__global__ void edge_aggr_l2(const int* __restrict__ ei, int E, int ET,
                             const float* __restrict__ h2,
                             const float* __restrict__ ex,
                             const float* __restrict__ denom,
                             float* __restrict__ out) {
    long long t = (long long)blockIdx.x * blockDim.x + threadIdx.x;
    if (t >= (long long)ET * 16) return;
    int e = (int)(t >> 4), c = (int)(t & 15);
    int s, d; edge_sd(ei, E, e, s, d);
    float alpha = ex[e] / (denom[d] + 1e-16f);
    atomicAdd(&out[(size_t)d * 16 + c], alpha * h2[(size_t)s * 16 + c]);
}

// out1 + b1 -> ELU -> f16 input for GEMM2
__global__ void bias_elu_f16(const float* __restrict__ acc, const float* __restrict__ b,
                             _Float16* __restrict__ xh, size_t n) {
    size_t t = (size_t)blockIdx.x * blockDim.x + threadIdx.x;
    if (t >= n) return;
    float v = acc[t] + b[t & 255];
    v = v > 0.0f ? v : (__expf(v) - 1.0f);
    xh[t] = (_Float16)v;
}

__global__ void finalize_k(const float* __restrict__ acc, const float* __restrict__ b2,
                           float* __restrict__ out, int Nn) {
    int t = blockIdx.x * blockDim.x + threadIdx.x;
    if (t >= Nn * 16) return;
    out[t] = acc[t] + b2[t & 15];
}

// ------------------------------------------------------------------
extern "C" void kernel_launch(void* const* d_in, const int* in_sizes, int n_in,
                              void* d_out, int out_size, void* d_ws, size_t ws_size,
                              hipStream_t stream) {
    const float* x        = (const float*)d_in[0];
    const int*   ei       = (const int*)d_in[1];   // [2,E] (src row, dst row)
    const float* W1       = (const float*)d_in[2];
    const float* att_src1 = (const float*)d_in[3];
    const float* att_dst1 = (const float*)d_in[4];
    const float* b1       = (const float*)d_in[5];
    const float* W2       = (const float*)d_in[6];
    const float* att_src2 = (const float*)d_in[7];
    const float* att_dst2 = (const float*)d_in[8];
    const float* b2       = (const float*)d_in[9];
    float* out = (float*)d_out;

    const int Nn = in_sizes[0] / 256;   // 50000
    const int E  = in_sizes[1] / 2;     // 800000
    const int ET = E + Nn;              // + self loops

    // workspace layout (256B aligned slices)
    char* ws = (char*)d_ws;
    size_t off = 0;
    auto alloc = [&](size_t bytes) -> void* {
        void* p = ws + off;
        off += (bytes + 255) & ~(size_t)255;
        return p;
    };
    _Float16* W1T   = (_Float16*)alloc((size_t)256 * 256 * 2);
    _Float16* W2T   = (_Float16*)alloc((size_t)256 * 16 * 2);
    _Float16* xh    = (_Float16*)alloc((size_t)Nn * 256 * 2);  // f16 x, reused as f16 elu(h)
    float*    h1    = (float*)alloc((size_t)Nn * 256 * 4);
    float*    as1   = (float*)alloc((size_t)Nn * 4 * 4);
    float*    ad1   = (float*)alloc((size_t)Nn * 4 * 4);
    unsigned* amax1 = (unsigned*)alloc((size_t)Nn * 4 * 4);
    float*    den1  = (float*)alloc((size_t)Nn * 4 * 4);
    float*    ex1   = (float*)alloc((size_t)ET * 4 * 4);
    float*    out1  = (float*)alloc((size_t)Nn * 256 * 4);
    float*    h2    = (float*)alloc((size_t)Nn * 16 * 4);
    float*    as2   = (float*)alloc((size_t)Nn * 4);
    float*    ad2   = (float*)alloc((size_t)Nn * 4);
    unsigned* amax2 = (unsigned*)alloc((size_t)Nn * 4);
    float*    den2  = (float*)alloc((size_t)Nn * 4);
    float*    ex2   = (float*)alloc((size_t)ET * 4);
    float*    out2  = (float*)alloc((size_t)Nn * 16 * 4);

    auto gridFor = [](size_t n, int blk) { return (int)((n + blk - 1) / blk); };

    // 0) weight transpose+cvt, x cvt
    cvt_transpose<<<gridFor(256 * 256, 256), 256, 0, stream>>>(W1, W1T, 256, 256);
    cvt_transpose<<<gridFor(256 * 16, 256), 256, 0, stream>>>(W2, W2T, 256, 16);
    cvt_f16<<<gridFor((size_t)Nn * 256, 256), 256, 0, stream>>>(x, xh, (size_t)Nn * 256);

    // 1) zero accumulators (encoded 0 == -inf sentinel for amax)
    zero_f32<<<gridFor((size_t)Nn * 4, 256), 256, 0, stream>>>((float*)amax1, (size_t)Nn * 4);
    zero_f32<<<gridFor((size_t)Nn * 4, 256), 256, 0, stream>>>(den1, (size_t)Nn * 4);
    zero_f32<<<gridFor((size_t)Nn * 256, 256), 256, 0, stream>>>(out1, (size_t)Nn * 256);
    zero_f32<<<gridFor((size_t)Nn, 256), 256, 0, stream>>>((float*)amax2, (size_t)Nn);
    zero_f32<<<gridFor((size_t)Nn, 256), 256, 0, stream>>>(den2, (size_t)Nn);
    zero_f32<<<gridFor((size_t)Nn * 16, 256), 256, 0, stream>>>(out2, (size_t)Nn * 16);

    // 2) layer 1: h1 = x @ W1  (wave per 16x64 strip: 4 wmma accumulators)
    {
        int waves = (Nn / 16) * (256 / 64);
        wmma_gemm<4><<<gridFor(waves, 8), 256, 0, stream>>>(xh, W1T, h1, Nn, 256, 256);
    }
    att_dots<<<gridFor((size_t)Nn * 4, 256), 256, 0, stream>>>(h1, att_src1, att_dst1,
                                                              as1, ad1, Nn, 4, 64);
    edge_max<<<gridFor(ET, 256), 256, 0, stream>>>(ei, E, ET, as1, ad1, amax1, 4);
    edge_expsum<<<gridFor(ET, 256), 256, 0, stream>>>(ei, E, ET, as1, ad1, amax1, ex1, den1, 4);
    edge_aggr_l1<<<gridFor(ET, 8), 256, 0, stream>>>(ei, E, ET, h1, ex1, den1, out1);

    // 3) ELU(out1 + b1) -> f16 (xh reused)
    bias_elu_f16<<<gridFor((size_t)Nn * 256, 256), 256, 0, stream>>>(out1, b1, xh,
                                                                     (size_t)Nn * 256);

    // 4) layer 2: h2 = elu_h @ W2  (N=16: single tile per wave)
    {
        int waves = (Nn / 16);
        wmma_gemm<1><<<gridFor(waves, 8), 256, 0, stream>>>(xh, W2T, h2, Nn, 16, 256);
    }
    att_dots<<<gridFor((size_t)Nn, 256), 256, 0, stream>>>(h2, att_src2, att_dst2,
                                                           as2, ad2, Nn, 1, 16);
    edge_max<<<gridFor(ET, 256), 256, 0, stream>>>(ei, E, ET, as2, ad2, amax2, 1);
    edge_expsum<<<gridFor(ET, 256), 256, 0, stream>>>(ei, E, ET, as2, ad2, amax2, ex2, den2, 1);
    edge_aggr_l2<<<gridFor((size_t)ET * 16, 256), 256, 0, stream>>>(ei, E, ET, h2, ex2, den2, out2);

    // 5) out = out2 + b2
    finalize_k<<<gridFor((size_t)Nn * 16, 256), 256, 0, stream>>>(out2, b2, out, Nn);
}
